// Condition9Trans_24799141167124
// MI455X (gfx1250) — matrix operand order repaired and checked by reading, not compiled, not run on hardware
//
#include <hip/hip_runtime.h>
#include <cstdint>

typedef __attribute__((ext_vector_type(16))) __bf16 v16bf;
typedef __attribute__((ext_vector_type(8)))  float  v8f;
typedef __attribute__((ext_vector_type(4)))  float  nf4;   // native vector for NT loads

#define FDIM 256
#define HID  256
#define HPAD 264          // padded k-stride (bf16 elems) -> 528B row pitch, breaks bank conflicts
#define ROWS_PER_BLOCK 128

union Frag {
  v16bf    v;
  uint32_t u[8];
  uint4    q[2];
};

__device__ __forceinline__ uint32_t f32_to_bf16_bits(float f) {
  uint32_t x = __float_as_uint(f);
  x += 0x7FFFu + ((x >> 16) & 1u);          // round-to-nearest-even
  return x >> 16;
}
__device__ __forceinline__ uint32_t pk_bf16(float lo, float hi) {
  uint32_t a = __float_as_uint(lo);
  a += 0x7FFFu + ((a >> 16) & 1u);
  uint32_t b = __float_as_uint(hi);
  b += 0x7FFFu + ((b >> 16) & 1u);
  return (a >> 16) | (b & 0xFFFF0000u);
}

// ---- prep: W1 [256][256] f32 -> W1^T [n][k] bf16 in ws ----
__global__ __launch_bounds__(256) void prep_w1(const float* __restrict__ W1,
                                               unsigned short* __restrict__ w1t) {
  const int n = blockIdx.x;          // 0..255 hidden unit
  const int k = threadIdx.x;         // 0..255 input feature
  w1t[n * FDIM + k] = (unsigned short)f32_to_bf16_bits(W1[k * HID + n]);
}

// ---- prep: W2 [256][9] f32 -> W2^T [16(pad)][256] bf16 in ws ----
__global__ __launch_bounds__(256) void prep_w2(const float* __restrict__ W2,
                                               unsigned short* __restrict__ w2t) {
  const int n = blockIdx.x;          // 0..15 (cols 9..15 zero)
  const int k = threadIdx.x;         // 0..255
  const float v = (n < 9) ? W2[k * 9 + n] : 0.0f;
  w2t[n * HID + k] = (unsigned short)f32_to_bf16_bits(v);
}

// Async (memory -> LDS, no VGPR round trip, ASYNCcnt-tracked) stage of one
// 16x256 bf16 W1^T slice. 32B per thread via two b128 async loads; the
// instruction offset applies to both the global and LDS addresses (ISA 08 §4.4).
__device__ __forceinline__ void async_stage_w1(const unsigned short* __restrict__ w1t,
                                               int nt, unsigned short* dstBase, int tid) {
  const int n = tid >> 4;            // 0..15 tile column
  const int c = tid & 15;            // 16-element k chunk
  const unsigned short* src = w1t + (size_t)(nt * 16 + n) * FDIM + c * 16;
  unsigned short* dst = dstBase + n * HPAD + c * 16;
  const uint32_t ldsoff = (uint32_t)(uintptr_t)dst;   // low 32 bits = LDS byte offset
  asm volatile(
      "global_load_async_to_lds_b128 %0, %1, off\n\t"
      "global_load_async_to_lds_b128 %0, %1, off offset:16"
      :: "v"(ldsoff), "v"(src) : "memory");
}

__device__ __forceinline__ void wait_asynccnt0() {
  asm volatile("s_wait_asynccnt 0x0" ::: "memory");
}

// forward-mode normalize over 3 tangent channels (matches _accp_normalize)
__device__ __forceinline__ void accp_normalize(const float raw[3], const float rawa[3][3],
                                               float t[3], float ta[3][3]) {
  const float ss  = raw[0]*raw[0] + raw[1]*raw[1] + raw[2]*raw[2];
  const float inv = 1.0f / sqrtf(ss);
  const float inv3 = inv * inv * inv;
#pragma unroll
  for (int a = 0; a < 3; ++a) t[a] = inv * raw[a];
#pragma unroll
  for (int k = 0; k < 3; ++k) {
    const float ssa  = 2.0f * (rawa[k][0]*raw[0] + rawa[k][1]*raw[1] + rawa[k][2]*raw[2]);
    const float inva = -0.5f * ssa * inv3;
#pragma unroll
    for (int a = 0; a < 3; ++a) ta[k][a] = inva * raw[a] + inv * rawa[k][a];
  }
}

__global__ __launch_bounds__(256) void fused_cond9(
    const float* __restrict__ rotation,        // [N][3][3]
    const float* __restrict__ feature,         // [N][256]
    const float* __restrict__ b1,              // [256]
    const float* __restrict__ b2,              // [9]
    const unsigned short* __restrict__ w1t,    // [256][256] bf16 (n-major)
    const unsigned short* __restrict__ w2t,    // [16][256]  bf16 (n-major)
    float* __restrict__ out,                   // [N*9] t_rot ++ [N] logdet
    int Ntot)
{
  __shared__ unsigned short sW1[2][16 * HPAD];     // double-buffered W1^T N-tile (bf16)
  __shared__ unsigned short sH[8 * 16 * HPAD];     // per-wave h tiles (bf16)
  __shared__ float          sMat[8 * 16 * 12];     // per-wave mat tiles (f32)

  const int tid  = threadIdx.x;
  const int lane = tid & 31;
  const int wv   = tid >> 5;      // wave 0..7
  const int l16  = lane & 15;
  const int lhi  = lane >> 4;     // 0/1 lane-group

  const int rowBase = blockIdx.x * ROWS_PER_BLOCK + wv * 16;
  const int myRow   = rowBase + l16;
  const int rc      = (myRow < Ntot) ? myRow : (Ntot - 1);

  // prefetch W1^T tile 0 into buffer 0 while we convert A fragments
  async_stage_w1(w1t, 0, sW1[0], tid);

  // ---- preload all A fragments for this wave's 16 feature rows (f32 -> bf16) ----
  // ISA 16-bit A 16x32 layout: lane M = l16 (group lhi), K = lhi*8 + (v%4)*2 + e + (v/4)*16
  Frag afr[8];
  {
    const float* fp = feature + (size_t)rc * FDIM;
#pragma unroll
    for (int kt = 0; kt < 8; ++kt) {
      const int kb = kt * 32 + lhi * 8;
      const nf4 a0 = __builtin_nontemporal_load((const nf4*)(fp + kb + 0));
      const nf4 a1 = __builtin_nontemporal_load((const nf4*)(fp + kb + 4));
      const nf4 a2 = __builtin_nontemporal_load((const nf4*)(fp + kb + 16));
      const nf4 a3 = __builtin_nontemporal_load((const nf4*)(fp + kb + 20));
      afr[kt].u[0] = pk_bf16(a0.x, a0.y);
      afr[kt].u[1] = pk_bf16(a0.z, a0.w);
      afr[kt].u[2] = pk_bf16(a1.x, a1.y);
      afr[kt].u[3] = pk_bf16(a1.z, a1.w);
      afr[kt].u[4] = pk_bf16(a2.x, a2.y);
      afr[kt].u[5] = pk_bf16(a2.z, a2.w);
      afr[kt].u[6] = pk_bf16(a3.x, a3.y);
      afr[kt].u[7] = pk_bf16(a3.z, a3.w);
    }
  }

  unsigned short* myH = sH + wv * (16 * HPAD);

  // ---- GEMM1: h = relu(feature @ W1 + b1), 16 N-tiles, double-buffered async weights ----
  for (int nt = 0; nt < 16; ++nt) {
    wait_asynccnt0();              // my async loads for tile nt have landed in LDS
    __syncthreads();               // everyone's have; previous buffer free for reuse
    if (nt < 15)
      async_stage_w1(w1t, nt + 1, sW1[(nt + 1) & 1], tid);   // overlap with WMMAs below

    const unsigned short* tile = sW1[nt & 1];
    const float b1v = b1[nt * 16 + l16];
    v8f acc = {0.f, 0.f, 0.f, 0.f, 0.f, 0.f, 0.f, 0.f};
#pragma unroll
    for (int kt = 0; kt < 8; ++kt) {
      // ISA 16-bit B 32x16 layout: lane N = l16, K = lhi*8 + (v%4)*2 + e + (v/4)*16
      Frag bfr;
      const uint4* bp = (const uint4*)(tile + l16 * HPAD + kt * 32 + lhi * 8);
      bfr.q[0] = bp[0];
      bfr.q[1] = bp[2];
      acc = __builtin_amdgcn_wmma_f32_16x16x32_bf16(false, afr[kt].v, false, bfr.v,
                                                    (short)0, acc, false, false);
    }
    // bias + relu, store h tile (bf16) in row-major [m][k] for GEMM2 A-fragments
#pragma unroll
    for (int v = 0; v < 8; ++v) {
      const float h = fmaxf(acc[v] + b1v, 0.0f);
      const int m = v + 8 * lhi;               // D layout: M = v + 8*lhi, N = l16
      myH[m * HPAD + nt * 16 + l16] = (unsigned short)f32_to_bf16_bits(h);
    }
  }

  // ---- GEMM2: mat = h @ W2 (N padded 9->16); A from own LDS h tile ----
  v8f acc2 = {0.f, 0.f, 0.f, 0.f, 0.f, 0.f, 0.f, 0.f};
#pragma unroll
  for (int kt = 0; kt < 8; ++kt) {
    Frag a2f, b2f;
    const uint4* ap = (const uint4*)(myH + l16 * HPAD + kt * 32 + lhi * 8);
    a2f.q[0] = ap[0];                          // K kb..kb+7
    a2f.q[1] = ap[2];                          // K kb+16..kb+23
    const uint4* bp = (const uint4*)(w2t + l16 * HID + kt * 32 + lhi * 8);
    b2f.q[0] = bp[0];
    b2f.q[1] = bp[2];
    acc2 = __builtin_amdgcn_wmma_f32_16x16x32_bf16(false, a2f.v, false, b2f.v,
                                                   (short)0, acc2, false, false);
  }

  // scatter mat tile so lane m owns point m's 3x3
  {
    float* mm = sMat + wv * 192;
    if (l16 < 9) {
#pragma unroll
      for (int v = 0; v < 8; ++v) {
        mm[(v + 8 * lhi) * 12 + l16] = acc2[v];
      }
    }
  }

  // ---- per-point epilogue: lanes 0..15 of each wave, one point each ----
  if (lhi == 0) {
    const int rr = rowBase + l16;
    if (rr < Ntot) {
      const float* mm = sMat + wv * 192 + l16 * 12;
      float M[3][3];
#pragma unroll
      for (int i = 0; i < 3; ++i)
#pragma unroll
        for (int j = 0; j < 3; ++j)
          M[i][j] = mm[i * 3 + j] + b2[i * 3 + j] + ((i == j) ? 1.0f : 0.0f);

      const float* rp = rotation + (size_t)rr * 9;
      float R[3][3];
#pragma unroll
      for (int i = 0; i < 3; ++i)
#pragma unroll
        for (int j = 0; j < 3; ++j)
          R[i][j] = __builtin_nontemporal_load(rp + i * 3 + j);

      float rm[3][3];
#pragma unroll
      for (int a = 0; a < 3; ++a)
#pragma unroll
        for (int c = 0; c < 3; ++c)
          rm[a][c] = M[a][0]*R[0][c] + M[a][1]*R[1][c] + M[a][2]*R[2][c];

      // c0 = rm[:,0], c1 = rm[:,1]; tangents from rm @ accp[k]
      float c0[3], c1[3], c0a[3][3], c1a[3][3];
#pragma unroll
      for (int a = 0; a < 3; ++a) {
        c0[a] = rm[a][0];
        c1[a] = rm[a][1];
        c0a[0][a] = -rm[a][1]; c0a[1][a] = -rm[a][2]; c0a[2][a] = 0.0f;
        c1a[0][a] =  rm[a][0]; c1a[1][a] = 0.0f;      c1a[2][a] = -rm[a][2];
      }

      float t0[3], t0a[3][3];
      accp_normalize(c0, c0a, t0, t0a);

      const float dot = t0[0]*c1[0] + t0[1]*c1[1] + t0[2]*c1[2];
      float dota[3], raw1[3], raw1a[3][3];
#pragma unroll
      for (int k = 0; k < 3; ++k)
        dota[k] = t0a[k][0]*c1[0] + t0a[k][1]*c1[1] + t0a[k][2]*c1[2]
                + t0[0]*c1a[k][0] + t0[1]*c1a[k][1] + t0[2]*c1a[k][2];
#pragma unroll
      for (int a = 0; a < 3; ++a) raw1[a] = c1[a] - dot * t0[a];
#pragma unroll
      for (int k = 0; k < 3; ++k)
#pragma unroll
        for (int a = 0; a < 3; ++a)
          raw1a[k][a] = c1a[k][a] - (dota[k]*t0[a] + dot*t0a[k][a]);

      float t1[3], t1a[3][3];
      accp_normalize(raw1, raw1a, t1, t1a);

      const int P1[3] = {1, 2, 0}, P2[3] = {2, 0, 1};
      float t2[3], t2a[3][3];
#pragma unroll
      for (int i = 0; i < 3; ++i)
        t2[i] = t0[P1[i]]*t1[P2[i]] - t0[P2[i]]*t1[P1[i]];
#pragma unroll
      for (int k = 0; k < 3; ++k)
#pragma unroll
        for (int i = 0; i < 3; ++i)
          t2a[k][i] = t0[P1[i]]*t1a[k][P2[i]] + t0a[k][P1[i]]*t1[P2[i]]
                    - t0[P2[i]]*t1a[k][P1[i]] - t0a[k][P2[i]]*t1[P1[i]];

      // V[k][m] = delta[k][(0,1),(0,2),(1,2)] with delta[k][a][c] = sum_b t_ba[k][a]*t_b[c]
      float V[3][3];
#pragma unroll
      for (int k = 0; k < 3; ++k) {
        V[k][0] = t0a[k][0]*t0[1] + t1a[k][0]*t1[1] + t2a[k][0]*t2[1];
        V[k][1] = t0a[k][0]*t0[2] + t1a[k][0]*t1[2] + t2a[k][0]*t2[2];
        V[k][2] = t0a[k][1]*t0[2] + t1a[k][1]*t1[2] + t2a[k][1]*t2[2];
      }
      const float d00 = V[1][1]*V[2][2] - V[1][2]*V[2][1];
      const float d01 = V[1][2]*V[2][0] - V[1][0]*V[2][2];
      const float d02 = V[1][0]*V[2][1] - V[1][1]*V[2][0];
      const float det = d00*V[0][0] + d01*V[0][1] + d02*V[0][2];
      const float ld  = logf(fabsf(det));

      float* op = out + (size_t)rr * 9;
#pragma unroll
      for (int a = 0; a < 3; ++a) {
        __builtin_nontemporal_store(t0[a], op + a * 3 + 0);
        __builtin_nontemporal_store(t1[a], op + a * 3 + 1);
        __builtin_nontemporal_store(t2[a], op + a * 3 + 2);
      }
      __builtin_nontemporal_store(ld, out + (size_t)Ntot * 9 + rr);
    }
  }
}

extern "C" void kernel_launch(void* const* d_in, const int* in_sizes, int n_in,
                              void* d_out, int out_size, void* d_ws, size_t ws_size,
                              hipStream_t stream) {
  (void)n_in; (void)out_size; (void)ws_size;
  const float* rotation = (const float*)d_in[0];
  const float* feature  = (const float*)d_in[1];
  const float* W1       = (const float*)d_in[2];
  const float* b1       = (const float*)d_in[3];
  const float* W2       = (const float*)d_in[4];
  const float* b2       = (const float*)d_in[5];
  float* out = (float*)d_out;
  const int Ntot = in_sizes[0] / 9;

  unsigned short* w1t = (unsigned short*)d_ws;       // 256*256 bf16 = 128 KB
  unsigned short* w2t = w1t + FDIM * HID;            // 16*256  bf16 =   8 KB

  prep_w1<<<HID, FDIM, 0, stream>>>(W1, w1t);
  prep_w2<<<16, HID, 0, stream>>>(W2, w2t);

  const int blocks = (Ntot + ROWS_PER_BLOCK - 1) / ROWS_PER_BLOCK;
  fused_cond9<<<blocks, 256, 0, stream>>>(rotation, feature, b1, b2, w1t, w2t, out, Ntot);
}
